// TransitionModel_90211493085974
// MI455X (gfx1250) — matrix operand order repaired and checked by reading, not compiled, not run on hardware
//
#include <hip/hip_runtime.h>
#include <hip/hip_bf16.h>

// ---------------------------------------------------------------------------
// Types for WMMA fragments (gfx1250, wave32)
// ---------------------------------------------------------------------------
typedef _Float16 half8 __attribute__((ext_vector_type(8)));
typedef _Float16 v16h  __attribute__((ext_vector_type(16)));
typedef float    v8f   __attribute__((ext_vector_type(8)));
typedef unsigned int uint4v __attribute__((ext_vector_type(4)));
typedef int          int4v  __attribute__((ext_vector_type(4)));
typedef int          int8v  __attribute__((ext_vector_type(8)));

__device__ __forceinline__ v8f wmma_f16(v16h a, v16h b, v8f c) {
    // v_wmma_f32_16x16x32_f16  (neg_a, A, neg_b, B, c_mod, C, reuse_a, reuse_b)
    return __builtin_amdgcn_wmma_f32_16x16x32_f16(false, a, false, b,
                                                  (short)0, c, false, false);
}

// Model constants
#define BATCH 8192
#define NN    32
#define KK    32
#define FLAT  1024
#define AA    128
#define HH    1024
#define CIN   1152   // FLAT + AA

// ---------------------------------------------------------------------------
// TDM: DMA a 2D strip of packed B fragments into LDS.
// Copy = 32 rows x 1024 bytes; row stride = KT*1024 bytes; data_size = 8B.
// D# layout per CDNA5 ISA.
// ---------------------------------------------------------------------------
__device__ __forceinline__ void tdm_load_bstrip(unsigned lds_addr,
                                                const _Float16* gptr,
                                                unsigned KT) {
    unsigned long long ga = (unsigned long long)(uintptr_t)gptr;
    uint4v g0;
    g0[0] = 1u;                                        // count=1 (valid user D#)
    g0[1] = lds_addr;                                  // LDS byte address
    g0[2] = (unsigned)(ga & 0xffffffffu);              // global_addr[31:0]
    g0[3] = (unsigned)((ga >> 32) & 0x1ffffffu)        // global_addr[56:32]
          | (2u << 30);                                // type = 2 ("image")
    const unsigned td0 = 128u;                         // tensor_dim0 (8B units)
    const unsigned td1 = 32u;                          // tensor_dim1 (rows)
    unsigned long long s0 = (unsigned long long)KT * 128ull; // dim0 stride, units
    int8v g1;
    g1[0] = (int)(3u << 16);                           // data_size=3 (8 bytes)
    g1[1] = (int)((td0 & 0xffffu) << 16);              // tensor_dim0 lo16
    g1[2] = (int)((td0 >> 16) | ((td1 & 0xffffu) << 16)); // td0 hi16 | td1 lo16
    g1[3] = (int)((td1 >> 16) | (128u << 16));         // td1 hi16 | tile_dim0=128
    g1[4] = (int)32u;                                  // tile_dim1=32, tile_dim2=0
    g1[5] = (int)(s0 & 0xffffffffu);                   // tensor_dim0_stride lo32
    g1[6] = (int)((s0 >> 32) & 0xffffu);               // stride0 hi16 | stride1 lo16=0
    g1[7] = 0;                                         // stride1 hi32 = 0
    int4v gz = {0, 0, 0, 0};
#if __clang_major__ >= 23
    int8v gz8 = {0, 0, 0, 0, 0, 0, 0, 0};
    __builtin_amdgcn_tensor_load_to_lds(g0, g1, gz, gz, gz8, 0);
#else
    __builtin_amdgcn_tensor_load_to_lds(g0, g1, gz, gz, 0);
#endif
}

// ---------------------------------------------------------------------------
// Pack f32 weight matrix W[N,K] (y = x @ W^T) into f16 B-fragment layout:
// tile (tn,tk): lane L holds col = tn*16 + L%16, k = tk*32 + (L/16)*8 + {0..7, 16..23},
// stored as 16 contiguous f16 per lane-slot (one 32B load in the GEMM).
// ---------------------------------------------------------------------------
__global__ void pack_w(const float* __restrict__ W, _Float16* __restrict__ out,
                       int N, int K, int nslots) {
    int t = blockIdx.x * 256 + threadIdx.x;
    if (t >= nslots) return;
    int lane = t & 31;
    int KT   = K >> 5;
    int tk   = (t >> 5) % KT;
    int tn   = (t >> 5) / KT;
    int col  = lane & 15;
    int half = lane >> 4;
    const float* src = W + (size_t)(tn * 16 + col) * K + tk * 32 + half * 8;
    _Float16*    dst = out + (size_t)t * 16;
#pragma unroll
    for (int j = 0; j < 16; ++j) dst[j] = (_Float16)src[j < 8 ? j : j + 8];
}

// ---------------------------------------------------------------------------
// QKV projection: [B*32, 32] x [32, 96] + bias, WMMA.
// Output column-major per batch: qkv_t[b][c][m] (c in 0..95, m in 0..31).
// Wave: 32 rows x 48 cols. grid = 2048 blocks x 256 thr (8 waves).
// ---------------------------------------------------------------------------
__global__ void qkv_kernel(const float* __restrict__ x,
                           const _Float16* __restrict__ bpack,
                           const float* __restrict__ bias,
                           _Float16* __restrict__ qkv_t) {
    int lane = threadIdx.x & 31;
    int wid  = threadIdx.x >> 5;
    int gid  = blockIdx.x * 8 + wid;
    int cg   = gid & 1;          // 0: cols 0..47, 1: cols 48..95
    int rg   = gid >> 1;         // 0..8191
    int r0   = rg * 32;
    int col16 = lane & 15, half = lane >> 4;

    v16h a[2];
#pragma unroll
    for (int it = 0; it < 2; ++it) {
        size_t row = (size_t)(r0 + it * 16 + col16);
        const float* ap = x + row * 32 + half * 8;
        float4 f0 = *(const float4*)(ap);
        float4 f1 = *(const float4*)(ap + 4);
        float4 f2 = *(const float4*)(ap + 16);
        float4 f3 = *(const float4*)(ap + 20);
        v16h av;
        av[0]=(_Float16)f0.x; av[1]=(_Float16)f0.y; av[2]=(_Float16)f0.z; av[3]=(_Float16)f0.w;
        av[4]=(_Float16)f1.x; av[5]=(_Float16)f1.y; av[6]=(_Float16)f1.z; av[7]=(_Float16)f1.w;
        av[8]=(_Float16)f2.x; av[9]=(_Float16)f2.y; av[10]=(_Float16)f2.z; av[11]=(_Float16)f2.w;
        av[12]=(_Float16)f3.x; av[13]=(_Float16)f3.y; av[14]=(_Float16)f3.z; av[15]=(_Float16)f3.w;
        a[it] = av;
    }

    v8f zero = {0.f,0.f,0.f,0.f,0.f,0.f,0.f,0.f};
    v8f acc[2][3];
#pragma unroll
    for (int it = 0; it < 2; ++it)
#pragma unroll
        for (int jt = 0; jt < 3; ++jt) acc[it][jt] = zero;

#pragma unroll
    for (int jt = 0; jt < 3; ++jt) {
        int ntile = cg * 3 + jt;
        v16h b = *(const v16h*)(bpack + (size_t)(ntile * 32 + lane) * 16);
#pragma unroll
        for (int it = 0; it < 2; ++it) acc[it][jt] = wmma_f16(a[it], b, acc[it][jt]);
    }

#pragma unroll
    for (int it = 0; it < 2; ++it)
#pragma unroll
        for (int jt = 0; jt < 3; ++jt) {
            int col  = cg * 48 + jt * 16 + col16;
            float bv = bias[col];
            int row0 = r0 + it * 16 + half * 8;   // first of 8 consecutive rows
            int bidx = row0 >> 5;
            int m0   = row0 & 31;
            half8 st;
#pragma unroll
            for (int v = 0; v < 8; ++v) st[v] = (_Float16)(acc[it][jt][v] + bv);
            *(half8*)(qkv_t + (size_t)bidx * 3072 + col * 32 + m0) = st;
        }
}

// ---------------------------------------------------------------------------
// Attention (num_heads=32, head_dim=1 -> rank-1 scores per head) + out_proj.
// One wave per batch element; lane = head (phase 1), lane = out col (phase 2).
// Writes ba[b] = [attn_belief(1024) | action_embed(128)] as f16.
// grid = B/4 blocks x 128 threads (4 waves).
// ---------------------------------------------------------------------------
__global__ void attn_kernel(const _Float16* __restrict__ qkv_t,
                            const float* __restrict__ opw,
                            const float* __restrict__ opb,
                            const float* __restrict__ action,
                            _Float16* __restrict__ ba) {
    __shared__ float o_lds[4][32][32];
    int lane = threadIdx.x & 31;
    int wid  = threadIdx.x >> 5;
    int b    = blockIdx.x * 4 + wid;

    const _Float16* base = qkv_t + (size_t)b * 3072;
    const _Float16* qp   = base + lane * 32;           // q column for head=lane
    float kk[32], vv[32];
#pragma unroll
    for (int j = 0; j < 32; ++j) {
        kk[j] = (float)base[(32 + lane) * 32 + j];
        vv[j] = (float)base[(64 + lane) * 32 + j];
    }
    for (int i = 0; i < 32; ++i) {
        float qi = (float)qp[i];
        float den = 0.f, num = 0.f;
#pragma unroll
        for (int j = 0; j < 32; ++j) {
            float e = __expf(qi * kk[j]);
            den += e;
            num = fmaf(e, vv[j], num);
        }
        o_lds[wid][i][lane] = num / den;
    }
    __syncthreads();

    // out_proj: out[i][c] = sum_h o[i][h] * opw[c][h] + opb[c]; c = lane
    float w[32];
#pragma unroll
    for (int h = 0; h < 32; ++h) w[h] = opw[lane * 32 + h];
    float bv = opb[lane];
    _Float16* bap = ba + (size_t)b * CIN;
    for (int i = 0; i < 32; ++i) {
        float acc = bv;
#pragma unroll
        for (int h = 0; h < 32; ++h) acc = fmaf(o_lds[wid][i][h], w[h], acc);
        bap[i * 32 + lane] = (_Float16)acc;
    }
    // append action embedding (f32 -> f16)
    const float* ae = action + (size_t)b * AA;
#pragma unroll
    for (int t = 0; t < 4; ++t)
        bap[FLAT + t * 32 + lane] = (_Float16)ae[t * 32 + lane];
}

// ---------------------------------------------------------------------------
// f16 WMMA GEMM with TDM-staged B: out[M=8192, N] = elu?(A[M,K] x Wpack + bias)
// Block = 8 waves = one 32-row group x 8 column-groups (512 cols of N=1024).
// Per k-step: wave 0 TDM-DMAs the next 32-tile B strip (32 KB) into the spare
// LDS buffer (tensorcnt), while all waves compute from the current buffer.
// Each wave: 32 rows x 64 cols (2x4 C tiles, 8 WMMA per 32-k step).
// grid = 512 blocks x 256 threads; N = 1024 fixed (colg=16, 2 blocks per rg).
// ---------------------------------------------------------------------------
__global__ void fc_gemm(const _Float16* __restrict__ A, int lda,
                        const _Float16* __restrict__ bpack,
                        const float* __restrict__ bias,
                        void* __restrict__ out,
                        int N, int K, int elu, int out_f16) {
    __shared__ _Float16 bbuf[2][32 * 512];   // 2 x 32 KB double buffer
    int lane = threadIdx.x & 31;
    int wid  = threadIdx.x >> 5;
    int cgBase = (blockIdx.x & 1) * 8;       // block covers col-groups cgBase..+7
    int rg     = blockIdx.x >> 1;            // row group (32 rows)
    int r0     = rg * 32;
    int col16 = lane & 15, half = lane >> 4;
    int KT    = K >> 5;
    int ntbase = cgBase * 4;                 // first of 32 staged n-tiles

    unsigned lds0 = (unsigned)(uintptr_t)(&bbuf[0][0]);
    unsigned lds1 = (unsigned)(uintptr_t)(&bbuf[1][0]);

    // prologue: stage k-step 0
    if (wid == 0) {
        tdm_load_bstrip(lds0, bpack + (size_t)(ntbase * KT + 0) * 512, (unsigned)KT);
        __builtin_amdgcn_s_wait_tensorcnt(0);
    }
    __syncthreads();

    v8f zero = {0.f,0.f,0.f,0.f,0.f,0.f,0.f,0.f};
    v8f acc[2][4];
#pragma unroll
    for (int it = 0; it < 2; ++it)
#pragma unroll
        for (int jt = 0; jt < 4; ++jt) acc[it][jt] = zero;

    for (int kt = 0; kt < KT; ++kt) {
        // kick off DMA of next strip into the other buffer
        if (wid == 0 && (kt + 1) < KT) {
            unsigned dst = ((kt + 1) & 1) ? lds1 : lds0;
            tdm_load_bstrip(dst, bpack + (size_t)(ntbase * KT + kt + 1) * 512,
                            (unsigned)KT);
        }

        v16h a[2];
#pragma unroll
        for (int it = 0; it < 2; ++it) {
            const _Float16* ap =
                A + (size_t)(r0 + it * 16 + col16) * lda + kt * 32 + half * 8;
            half8 lo = *(const half8*)(ap);
            half8 hi = *(const half8*)(ap + 16);
            a[it] = __builtin_shufflevector(lo, hi,
                     0,1,2,3,4,5,6,7,8,9,10,11,12,13,14,15);
        }
        const _Float16* bsrc = &bbuf[kt & 1][0];
#pragma unroll
        for (int jt = 0; jt < 4; ++jt) {
            int ltile = wid * 4 + jt;        // local tile index in staged strip
            v16h b = *(const v16h*)(bsrc + (size_t)(ltile * 32 + lane) * 16);
#pragma unroll
            for (int it = 0; it < 2; ++it)
                acc[it][jt] = wmma_f16(a[it], b, acc[it][jt]);
        }

        if (wid == 0) __builtin_amdgcn_s_wait_tensorcnt(0);
        __syncthreads();
    }

#pragma unroll
    for (int it = 0; it < 2; ++it)
#pragma unroll
        for (int jt = 0; jt < 4; ++jt) {
            int col  = (cgBase + wid) * 64 + jt * 16 + col16;
            float bv = bias[col];
            int rowb = r0 + it * 16 + half * 8;
#pragma unroll
            for (int v = 0; v < 8; ++v) {
                float x = acc[it][jt][v] + bv;
                if (elu) x = x > 0.f ? x : (__expf(x) - 1.f);
                if (out_f16)
                    ((_Float16*)out)[(size_t)(rowb + v) * N + col] = (_Float16)x;
                else
                    ((float*)out)[(size_t)(rowb + v) * N + col] = x;
            }
        }
}

// ---------------------------------------------------------------------------
// Gumbel-softmax over last dim (K=32). lane = category, wave-wide reduce.
// grid = B*N/8 blocks x 256 threads.
// ---------------------------------------------------------------------------
__global__ void gumbel_kernel(const float* __restrict__ logits,
                              const float* __restrict__ u,
                              const float* __restrict__ temp,
                              float* __restrict__ out, int nrows) {
    int lane = threadIdx.x & 31;
    int wid  = threadIdx.x >> 5;
    int row  = blockIdx.x * 8 + wid;
    if (row >= nrows) return;
    float lg = logits[(size_t)row * 32 + lane];
    float uu = u[(size_t)row * 32 + lane];
    float g  = -__logf(-__logf(uu + 1e-20f) + 1e-20f);
    float y  = (lg + g) / temp[0];
    float m  = y;
#pragma unroll
    for (int off = 16; off > 0; off >>= 1) m = fmaxf(m, __shfl_xor(m, off, 32));
    float e = __expf(y - m);
    float s = e;
#pragma unroll
    for (int off = 16; off > 0; off >>= 1) s += __shfl_xor(s, off, 32);
    out[(size_t)row * 32 + lane] = e / s;
}

// ---------------------------------------------------------------------------
// Workspace layout (bytes):
//   0        : in_proj pack   (6144)
//   6144     : fc1 pack       (2359296)
//   2365440  : fc2 pack       (2097152)
//   4462592  : fc3 pack       (2097152)
//   6559744  : ba  f16        (18874368)
//   25434112 : scratch 50.3MB : qkv_t (phase A), then h1 / h2 (phase B)
// total ~75.8 MB
// ---------------------------------------------------------------------------
extern "C" void kernel_launch(void* const* d_in, const int* in_sizes, int n_in,
                              void* d_out, int out_size, void* d_ws, size_t ws_size,
                              hipStream_t stream) {
    (void)in_sizes; (void)n_in; (void)out_size; (void)ws_size;
    const float* latent = (const float*)d_in[0];
    const float* action = (const float*)d_in[1];
    const float* gum    = (const float*)d_in[2];
    const float* temp   = (const float*)d_in[3];
    const float* inpw   = (const float*)d_in[4];
    const float* inpb   = (const float*)d_in[5];
    const float* opw    = (const float*)d_in[6];
    const float* opb    = (const float*)d_in[7];
    const float* fc1w   = (const float*)d_in[8];
    const float* fc1b   = (const float*)d_in[9];
    const float* fc2w   = (const float*)d_in[10];
    const float* fc2b   = (const float*)d_in[11];
    const float* fc3w   = (const float*)d_in[12];
    const float* fc3b   = (const float*)d_in[13];

    char* ws = (char*)d_ws;
    _Float16* p_inproj = (_Float16*)(ws);
    _Float16* p_fc1    = (_Float16*)(ws + 6144);
    _Float16* p_fc2    = (_Float16*)(ws + 2365440);
    _Float16* p_fc3    = (_Float16*)(ws + 4462592);
    _Float16* ba       = (_Float16*)(ws + 6559744);
    _Float16* qkv      = (_Float16*)(ws + 25434112);
    _Float16* h1       = (_Float16*)(ws + 25434112);
    _Float16* h2       = (_Float16*)(ws + 25434112 + 16777216);

    float* pred   = (float*)d_out;
    float* logits = (float*)d_out + (size_t)BATCH * FLAT;

    // 1) pack weights to f16 B-fragment layout
    pack_w<<<1,   256, 0, stream>>>(inpw, p_inproj, 96,   32,   192);
    pack_w<<<288, 256, 0, stream>>>(fc1w, p_fc1,    HH,   CIN,  73728);
    pack_w<<<256, 256, 0, stream>>>(fc2w, p_fc2,    HH,   HH,   65536);
    pack_w<<<256, 256, 0, stream>>>(fc3w, p_fc3,    FLAT, HH,   65536);

    // 2) qkv projection (WMMA), column-major per batch
    qkv_kernel<<<2048, 256, 0, stream>>>(latent, p_inproj, inpb, qkv);

    // 3) rank-1 attention + out_proj + concat action -> ba (f16)
    attn_kernel<<<BATCH / 4, 128, 0, stream>>>(qkv, opw, opb, action, ba);

    // 4) MLP (WMMA + TDM B staging): fc1+ELU, fc2+ELU, fc3 -> logits (f32)
    fc_gemm<<<512, 256, 0, stream>>>(ba, CIN, p_fc1, fc1b, (void*)h1,
                                     HH, CIN, 1, 1);
    fc_gemm<<<512, 256, 0, stream>>>(h1, HH, p_fc2, fc2b, (void*)h2,
                                     HH, HH, 1, 1);
    fc_gemm<<<512, 256, 0, stream>>>(h2, HH, p_fc3, fc3b, (void*)logits,
                                     FLAT, HH, 0, 0);

    // 5) gumbel-softmax -> pred (first output)
    gumbel_kernel<<<(BATCH * NN) / 8, 256, 0, stream>>>(logits, gum, temp,
                                                        pred, BATCH * NN);
}
